// CrossAttentionSkip_17497696764504
// MI455X (gfx1250) — matrix operand order repaired and checked by reading, not compiled, not run on hardware
//
#include <hip/hip_runtime.h>
#include <hip/hip_bf16.h>
#include <math.h>

#define NTOK 4096
#define BATCH 2
#define NHEADS 4
#define HDIM 32

typedef __attribute__((ext_vector_type(16))) _Float16 v16h;
typedef __attribute__((ext_vector_type(8)))  _Float16 v8h;
typedef __attribute__((ext_vector_type(8)))  float    v8f;

// ---- WMMA helpers -------------------------------------------------------
__device__ __forceinline__ v8f wmma16(v16h a, v16h b, v8f c) {
  return __builtin_amdgcn_wmma_f32_16x16x32_f16(false, a, false, b,
                                                (short)0, c, false, false);
}

// A-operand (16x32, M x K), row-major source, ld in halves.
// lanes 0-15: M=lane, K=0..15; lanes 16-31: K=16..31
__device__ __forceinline__ v16h load_tileA(const _Float16* base, int ld) {
  const int lane = threadIdx.x & 31;
  const int m = lane & 15, sel = lane >> 4;
  const _Float16* p = base + (size_t)m * ld + sel * 16;
  v8h lo = *(const v8h*)(p);
  v8h hi = *(const v8h*)(p + 8);
  v16h r;
#pragma unroll
  for (int i = 0; i < 8; ++i) { r[i] = lo[i]; r[i + 8] = hi[i]; }
  return r;
}

// B-operand (32x16, K x N), columns contiguous in memory, ld in halves.
// lanes 0-15: N=lane, K=0..15; lanes 16-31: K=16..31
__device__ __forceinline__ v16h load_tileB(const _Float16* base, int ld) {
  const int lane = threadIdx.x & 31;
  const int n = lane & 15, sel = lane >> 4;
  const _Float16* p = base + (size_t)n * ld + sel * 16;
  v8h lo = *(const v8h*)(p);
  v8h hi = *(const v8h*)(p + 8);
  v16h r;
#pragma unroll
  for (int i = 0; i < 8; ++i) { r[i] = lo[i]; r[i + 8] = hi[i]; }
  return r;
}

// ---- CDNA5 async global->LDS copy (ASYNCcnt path, ISA 15.18.3 op 98) ----
__device__ __forceinline__ void async_copy_b128(unsigned lds_byte_off,
                                                const _Float16* gptr) {
  asm volatile("global_load_async_to_lds_b128 %0, %1, off"
               :: "v"(lds_byte_off), "v"(gptr) : "memory");
}
__device__ __forceinline__ void wait_async0() {
  asm volatile("s_wait_asynccnt 0x0" ::: "memory");
}

// ---- weight f32 -> f16 --------------------------------------------------
__global__ void cvt_f16(const float* __restrict__ src, _Float16* __restrict__ dst, int n) {
  int i = blockIdx.x * blockDim.x + threadIdx.x;
  if (i < n) dst[i] = (_Float16)src[i];
}

// ---- kernel 1: layernorm + QKV projections ------------------------------
__global__ void ln_qkv_kernel(
    const float* __restrict__ enc, const float* __restrict__ dec,
    const _Float16* __restrict__ WqH, const float* __restrict__ bq,
    const _Float16* __restrict__ WkH, const float* __restrict__ bk,
    const _Float16* __restrict__ WvH, const float* __restrict__ bv,
    const float* __restrict__ g_enc, const float* __restrict__ b_enc,
    const float* __restrict__ g_dec, const float* __restrict__ b_dec,
    float* __restrict__ dec_ln,          // [B,N,128] f32 (residual)
    _Float16* __restrict__ Qh,           // [B,N,128]
    _Float16* __restrict__ Kh,           // [B,N,128]
    _Float16* __restrict__ Vt)           // [B,128,N] (transposed)
{
  __shared__ __align__(16) float    s_dec32[16][128];
  __shared__ __align__(16) float    s_enc32[16][128];
  __shared__ __align__(16) _Float16 s_dec[16][128];
  __shared__ __align__(16) _Float16 s_enc[16][128];
  __shared__ float s_mu[2][16], s_rs[2][16];

  const int n0 = blockIdx.x * 16;
  const int b  = blockIdx.y;
  const int t  = threadIdx.x;
  const int r  = t & 15;

  for (int c = t >> 4; c < 128; c += 8) {
    s_dec32[r][c] = dec[((size_t)b * 128 + c) * NTOK + n0 + r];
    s_enc32[r][c] = enc[((size_t)b * 128 + c) * NTOK + n0 + r];
  }
  __syncthreads();
  if (t < 32) {
    const float* row = (t < 16) ? s_dec32[t] : s_enc32[t - 16];
    float mu = 0.f;
    for (int c = 0; c < 128; ++c) mu += row[c];
    mu *= (1.f / 128.f);
    float var = 0.f;
    for (int c = 0; c < 128; ++c) { float d = row[c] - mu; var += d * d; }
    var *= (1.f / 128.f);
    s_mu[t >> 4][t & 15] = mu;
    s_rs[t >> 4][t & 15] = rsqrtf(var + 1e-5f);
  }
  __syncthreads();
  for (int c = t >> 4; c < 128; c += 8) {
    float xd = (s_dec32[r][c] - s_mu[0][r]) * s_rs[0][r] * g_dec[c] + b_dec[c];
    float xe = (s_enc32[r][c] - s_mu[1][r]) * s_rs[1][r] * g_enc[c] + b_enc[c];
    s_dec[r][c] = (_Float16)xd;
    s_enc[r][c] = (_Float16)xe;
    dec_ln[((size_t)b * NTOK + n0 + r) * 128 + c] = xd;
  }
  __syncthreads();

  const int wv = t >> 5, lane = t & 31;
  const int col = lane & 15, sel = lane >> 4, row8 = sel * 8;
  for (int ci = 0; ci < 2; ++ci) {
    const int o0 = (wv * 2 + ci) * 16;
    v8f accq, acck, accv;
    const float vbq = bq[o0 + col], vbk = bk[o0 + col], vbv = bv[o0 + col];
#pragma unroll
    for (int i = 0; i < 8; ++i) { accq[i] = vbq; acck[i] = vbk; accv[i] = vbv; }
#pragma unroll
    for (int kk = 0; kk < 4; ++kk) {
      v16h ad  = load_tileA(&s_dec[0][0] + kk * 32, 128);
      v16h ae  = load_tileA(&s_enc[0][0] + kk * 32, 128);
      v16h bqm = load_tileB(WqH + (size_t)o0 * 128 + kk * 32, 128);
      v16h bkm = load_tileB(WkH + (size_t)o0 * 128 + kk * 32, 128);
      v16h bvm = load_tileB(WvH + (size_t)o0 * 128 + kk * 32, 128);
      accq = wmma16(ad, bqm, accq);
      acck = wmma16(ae, bkm, acck);
      accv = wmma16(ae, bvm, accv);
    }
#pragma unroll
    for (int j = 0; j < 8; ++j) {
      const int tok = n0 + row8 + j;
      Qh[((size_t)b * NTOK + tok) * 128 + o0 + col] = (_Float16)accq[j];
      Kh[((size_t)b * NTOK + tok) * 128 + o0 + col] = (_Float16)acck[j];
      Vt[((size_t)b * 128 + o0 + col) * NTOK + tok] = (_Float16)accv[j];
    }
  }
}

// ---- kernel 2: flash attention with async-LDS staged K/V ----------------
// grid (N/256, heads, B), block 256 (8 waves); each wave owns TWO 16-row
// q tiles. K/V chunks (32 k x 32 d) are staged in LDS once per block via
// global_load_async_to_lds_b128 with double buffering.
__device__ __forceinline__ v16h softmax_pack(const v8f& st0, const v8f& st1,
                                             float& mrun, float& ssum,
                                             v8f& ota, v8f& otb) {
  const int sel = (threadIdx.x >> 4) & 1;
  const float scale = 0.17677669529663687f;  // 1/sqrt(32)
  float cm = -3.0e38f;
#pragma unroll
  for (int j = 0; j < 8; ++j) { cm = fmaxf(cm, st0[j]); cm = fmaxf(cm, st1[j]); }
  cm = fmaxf(cm, __shfl_xor(cm, 16, 32));
  const float mnew = fmaxf(mrun, cm);
  const float a    = scale * mnew;
  const float rsc  = __expf(scale * mrun - a);
  mrun = mnew;
  float p0[8], p1[8], lsum = 0.f;
#pragma unroll
  for (int j = 0; j < 8; ++j) {
    p0[j] = __expf(st0[j] * scale - a);
    p1[j] = __expf(st1[j] * scale - a);
    lsum += p0[j] + p1[j];
  }
  ssum = ssum * rsc + lsum;
#pragma unroll
  for (int j = 0; j < 8; ++j) { ota[j] *= rsc; otb[j] *= rsc; }
  v16h pt;
#pragma unroll
  for (int j = 0; j < 8; ++j) {
    const float x0 = __shfl_xor(p0[j], 16, 32);
    const float x1 = __shfl_xor(p1[j], 16, 32);
    pt[j]     = (_Float16)((sel == 0) ? p0[j] : x1);   // K = sel*16 + j
    pt[j + 8] = (_Float16)((sel == 0) ? x0    : p1[j]); // K = sel*16 + 8 + j
  }
  return pt;
}

__global__ void __launch_bounds__(256) attn_kernel(
    const _Float16* __restrict__ Qh,
    const _Float16* __restrict__ Kh,
    const _Float16* __restrict__ Vt,
    _Float16* __restrict__ attnO)        // [B,N,128]
{
  __shared__ __align__(16) _Float16 s_k[2][32][32];   // [buf][k][d]
  __shared__ __align__(16) _Float16 s_v[2][32][32];   // [buf][d][k]

  const int t    = threadIdx.x;
  const int wv   = t >> 5, lane = t & 31;
  const int h    = blockIdx.y;
  const int b    = blockIdx.z;
  const int q0   = blockIdx.x * 256 + wv * 16;
  const int q1   = q0 + 128;
  const int col  = lane & 15, sel = lane >> 4;

  const _Float16* Kb = Kh + (size_t)b * NTOK * 128 + h * HDIM;
  const _Float16* Vb = Vt + ((size_t)b * 128 + h * HDIM) * NTOK;

  // per-thread async-copy assignment: one b128 per chunk
  const int  crow = (t & 127) >> 2;          // 0..31
  const int  cseg = (t & 3) * 8;             // half offset within row
  const bool isK  = t < 128;
  const _Float16* gsrc0 = isK ? (Kb + (size_t)crow * 128 + cseg)
                              : (Vb + (size_t)crow * NTOK + cseg);
  const size_t    gstep = isK ? (size_t)32 * 128 : (size_t)32;
  const unsigned  lds0  = isK
      ? (unsigned)(size_t)&s_k[0][crow][cseg]
      : (unsigned)(size_t)&s_v[0][crow][cseg];

  // prologue: stage chunk 0
  async_copy_b128(lds0, gsrc0);

  // Q^T B-operands, resident across the whole k sweep
  const v16h bq0 = load_tileB(Qh + ((size_t)b * NTOK + q0) * 128 + h * HDIM, 128);
  const v16h bq1 = load_tileB(Qh + ((size_t)b * NTOK + q1) * 128 + h * HDIM, 128);

  v8f ot00 = {}, ot01 = {}, ot10 = {}, ot11 = {};
  float m0 = -3.0e38f, m1 = -3.0e38f, s0 = 0.f, s1 = 0.f;

  const int iters = NTOK / 32;
  for (int it = 0; it < iters; ++it) {
    wait_async0();              // this wave's copy of chunk `it` landed
    __syncthreads();            // all copies visible; prev buffer free
    if (it + 1 < iters)
      async_copy_b128(lds0 + (((it + 1) & 1) << 11), gsrc0 + (size_t)(it + 1) * gstep);

    const _Float16* kb = &s_k[it & 1][0][0];
    const _Float16* vb = &s_v[it & 1][0][0];
    const v16h ak0 = load_tileA(kb, 32);
    const v16h ak1 = load_tileA(kb + 16 * 32, 32);

    v8f st00 = {}, st01 = {}, st10 = {}, st11 = {};
    st00 = wmma16(ak0, bq0, st00);   // rows k..k+15,  cols q-group0
    st01 = wmma16(ak1, bq0, st01);   // rows k+16..31, cols q-group0
    st10 = wmma16(ak0, bq1, st10);
    st11 = wmma16(ak1, bq1, st11);

    const v16h pt0 = softmax_pack(st00, st01, m0, s0, ot00, ot01);
    const v16h pt1 = softmax_pack(st10, st11, m1, s1, ot10, ot11);

    const v16h av0 = load_tileA(vb, 32);            // d rows 0..15
    const v16h av1 = load_tileA(vb + 16 * 32, 32);  // d rows 16..31
    ot00 = wmma16(av0, pt0, ot00);
    ot01 = wmma16(av1, pt0, ot01);
    ot10 = wmma16(av0, pt1, ot10);
    ot11 = wmma16(av1, pt1, ot11);
  }

  s0 += __shfl_xor(s0, 16, 32);
  s1 += __shfl_xor(s1, 16, 32);
  const float inv0 = 1.f / s0, inv1 = 1.f / s1;
  _Float16* O0 = attnO + ((size_t)b * NTOK + q0) * 128 + h * HDIM;
  _Float16* O1 = attnO + ((size_t)b * NTOK + q1) * 128 + h * HDIM;
#pragma unroll
  for (int j = 0; j < 8; ++j) {
    const int d = j + 8 * sel;
    O0[(size_t)col * 128 + d]      = (_Float16)(ot00[j] * inv0);
    O0[(size_t)col * 128 + d + 16] = (_Float16)(ot01[j] * inv0);
    O1[(size_t)col * 128 + d]      = (_Float16)(ot10[j] * inv1);
    O1[(size_t)col * 128 + d + 16] = (_Float16)(ot11[j] * inv1);
  }
}

// ---- kernel 3: O-proj + residual + LN + FFN + residual ------------------
__global__ void oproj_ffn_kernel(
    const _Float16* __restrict__ attnO,
    const _Float16* __restrict__ WoH, const float* __restrict__ bo,
    const float* __restrict__ dec_ln,
    const float* __restrict__ g_out, const float* __restrict__ b_out,
    const _Float16* __restrict__ W1H, const float* __restrict__ b1,
    const _Float16* __restrict__ W2H, const float* __restrict__ b2,
    float* __restrict__ out)             // [B,128,N]
{
  __shared__ __align__(16) float    s_out[16][128];
  __shared__ __align__(16) _Float16 s_h[16][128];
  __shared__ __align__(16) _Float16 s_g[16][512];
  __shared__ float s_mu[16], s_rs[16];

  const int n0 = blockIdx.x * 16;
  const int b  = blockIdx.y;
  const int t  = threadIdx.x;
  const int wv = t >> 5, lane = t & 31;
  const int col = lane & 15, sel = lane >> 4, row8 = sel * 8;

  const _Float16* Abase = attnO + ((size_t)b * NTOK + n0) * 128;

  for (int ci = 0; ci < 2; ++ci) {
    const int o0 = (wv * 2 + ci) * 16;
    v8f acc; const float bb = bo[o0 + col];
#pragma unroll
    for (int i = 0; i < 8; ++i) acc[i] = bb;
#pragma unroll
    for (int kk = 0; kk < 4; ++kk) {
      v16h a = load_tileA(Abase + kk * 32, 128);
      v16h w = load_tileB(WoH + (size_t)o0 * 128 + kk * 32, 128);
      acc = wmma16(a, w, acc);
    }
#pragma unroll
    for (int j = 0; j < 8; ++j) {
      const int m = row8 + j;
      s_out[m][o0 + col] =
          acc[j] + dec_ln[((size_t)b * NTOK + n0 + m) * 128 + o0 + col];
    }
  }
  __syncthreads();
  if (t < 16) {
    float mu = 0.f;
    for (int c = 0; c < 128; ++c) mu += s_out[t][c];
    mu *= (1.f / 128.f);
    float var = 0.f;
    for (int c = 0; c < 128; ++c) { float d = s_out[t][c] - mu; var += d * d; }
    var *= (1.f / 128.f);
    s_mu[t] = mu; s_rs[t] = rsqrtf(var + 1e-5f);
  }
  __syncthreads();
  {
    const int r = t & 15;
    for (int c = t >> 4; c < 128; c += 8)
      s_h[r][c] = (_Float16)((s_out[r][c] - s_mu[r]) * s_rs[r] * g_out[c] + b_out[c]);
  }
  __syncthreads();

  for (int ci = 0; ci < 8; ++ci) {
    const int o0 = (wv * 8 + ci) * 16;
    v8f acc; const float bb = b1[o0 + col];
#pragma unroll
    for (int i = 0; i < 8; ++i) acc[i] = bb;
#pragma unroll
    for (int kk = 0; kk < 4; ++kk) {
      v16h a = load_tileA(&s_h[0][0] + kk * 32, 128);
      v16h w = load_tileB(W1H + (size_t)o0 * 128 + kk * 32, 128);
      acc = wmma16(a, w, acc);
    }
#pragma unroll
    for (int j = 0; j < 8; ++j) {
      const float x = acc[j];
      s_g[row8 + j][o0 + col] =
          (_Float16)(0.5f * x * (1.f + erff(x * 0.7071067811865475f)));
    }
  }
  __syncthreads();

  for (int ci = 0; ci < 2; ++ci) {
    const int o0 = (wv * 2 + ci) * 16;
    v8f acc; const float bb = b2[o0 + col];
#pragma unroll
    for (int i = 0; i < 8; ++i) acc[i] = bb;
#pragma unroll
    for (int kk = 0; kk < 16; ++kk) {
      v16h a = load_tileA(&s_g[0][0] + kk * 32, 512);
      v16h w = load_tileB(W2H + (size_t)o0 * 512 + kk * 32, 512);
      acc = wmma16(a, w, acc);
    }
#pragma unroll
    for (int j = 0; j < 8; ++j) {
      const int m = row8 + j;
      out[((size_t)b * 128 + o0 + col) * NTOK + n0 + m] = acc[j] + s_out[m][o0 + col];
    }
  }
}

// ---- launch --------------------------------------------------------------
extern "C" void kernel_launch(void* const* d_in, const int* in_sizes, int n_in,
                              void* d_out, int out_size, void* d_ws, size_t ws_size,
                              hipStream_t stream) {
  (void)in_sizes; (void)n_in; (void)out_size; (void)ws_size;
  const float* enc = (const float*)d_in[0];
  const float* dec = (const float*)d_in[1];
  const float* Wq  = (const float*)d_in[2];  const float* bq  = (const float*)d_in[3];
  const float* Wk  = (const float*)d_in[4];  const float* bk  = (const float*)d_in[5];
  const float* Wv  = (const float*)d_in[6];  const float* bv  = (const float*)d_in[7];
  const float* Wo  = (const float*)d_in[8];  const float* bo  = (const float*)d_in[9];
  const float* g_enc = (const float*)d_in[10]; const float* b_enc = (const float*)d_in[11];
  const float* g_dec = (const float*)d_in[12]; const float* b_dec = (const float*)d_in[13];
  const float* g_out = (const float*)d_in[14]; const float* b_out = (const float*)d_in[15];
  const float* W1  = (const float*)d_in[16]; const float* b1 = (const float*)d_in[17];
  const float* W2  = (const float*)d_in[18]; const float* b2 = (const float*)d_in[19];
  float* out = (float*)d_out;

  char* ws = (char*)d_ws;
  size_t off = 0;
  auto alloc = [&](size_t bytes) -> void* {
    void* p = ws + off;
    off = (off + bytes + 255) & ~(size_t)255;
    return p;
  };
  _Float16* WqH = (_Float16*)alloc((size_t)128 * 128 * 2);
  _Float16* WkH = (_Float16*)alloc((size_t)128 * 128 * 2);
  _Float16* WvH = (_Float16*)alloc((size_t)128 * 128 * 2);
  _Float16* WoH = (_Float16*)alloc((size_t)128 * 128 * 2);
  _Float16* W1H = (_Float16*)alloc((size_t)512 * 128 * 2);
  _Float16* W2H = (_Float16*)alloc((size_t)128 * 512 * 2);
  float*    dln = (float*)   alloc((size_t)BATCH * NTOK * 128 * 4);
  _Float16* Qh  = (_Float16*)alloc((size_t)BATCH * NTOK * 128 * 2);
  _Float16* Kh  = (_Float16*)alloc((size_t)BATCH * NTOK * 128 * 2);
  _Float16* Vt  = (_Float16*)alloc((size_t)BATCH * NTOK * 128 * 2);
  _Float16* Oh  = (_Float16*)alloc((size_t)BATCH * NTOK * 128 * 2);

  const int nw = 128 * 128, nf1 = 512 * 128, nf2 = 128 * 512;
  cvt_f16<<<(nw  + 255) / 256, 256, 0, stream>>>(Wq, WqH, nw);
  cvt_f16<<<(nw  + 255) / 256, 256, 0, stream>>>(Wk, WkH, nw);
  cvt_f16<<<(nw  + 255) / 256, 256, 0, stream>>>(Wv, WvH, nw);
  cvt_f16<<<(nw  + 255) / 256, 256, 0, stream>>>(Wo, WoH, nw);
  cvt_f16<<<(nf1 + 255) / 256, 256, 0, stream>>>(W1, W1H, nf1);
  cvt_f16<<<(nf2 + 255) / 256, 256, 0, stream>>>(W2, W2H, nf2);

  ln_qkv_kernel<<<dim3(NTOK / 16, BATCH), 128, 0, stream>>>(
      enc, dec, WqH, bq, WkH, bk, WvH, bv,
      g_enc, b_enc, g_dec, b_dec, dln, Qh, Kh, Vt);

  attn_kernel<<<dim3(NTOK / 256, NHEADS, BATCH), 256, 0, stream>>>(Qh, Kh, Vt, Oh);

  oproj_ffn_kernel<<<dim3(NTOK / 16, BATCH), 128, 0, stream>>>(
      Oh, WoH, bo, dln, g_out, b_out, W1H, b1, W2H, b2, out);
}
